// GCN_50912542327362
// MI455X (gfx1250) — compile-verified, matrix operand order checked by reading, not run on hardware
//
#include <hip/hip_runtime.h>
#include <math.h>

#define N_NODES 51200
#define N_ROI   400
#define HID     64
#define N_EDGES 1638400
#define N_GRAPHS 128
#define KFC1    (N_ROI * HID)   /* 25600 */
#define NFC1    256
#define KSPLIT  8               /* split-K factor for fc1 */
#define KSTEPS  (KFC1 / 32)     /* 800 */
#define NEG_SLOPE 0.01f

typedef __attribute__((ext_vector_type(16))) _Float16 v16h;
typedef __attribute__((ext_vector_type(8)))  _Float16 v8h;
typedef __attribute__((ext_vector_type(8)))  float    v8f;
typedef __attribute__((ext_vector_type(4)))  float    v4f;

static __device__ __forceinline__ v16h hcombine(v8h a, v8h b) {
    return __builtin_shufflevector(a, b, 0, 1, 2, 3, 4, 5, 6, 7,
                                         8, 9, 10, 11, 12, 13, 14, 15);
}

// ---------------------------------------------------------------- norm prep
__global__ void k_deg_init(float* deg) {
    int i = blockIdx.x * 256 + threadIdx.x;
    if (i < N_NODES) deg[i] = 1.0f;            // self-loop weight
}

__global__ void k_deg_edges(const int* __restrict__ dst,
                            const float* __restrict__ w,
                            float* __restrict__ deg) {
    int e = blockIdx.x * 256 + threadIdx.x;
    if (e >= N_EDGES) return;
    if (e + 4096 < N_EDGES) __builtin_prefetch(&w[e + 4096], 0, 0);
    atomicAdd(&deg[dst[e]], w[e]);
}

__global__ void k_dis(float* deg) {            // in-place deg -> D^-1/2
    int i = blockIdx.x * 256 + threadIdx.x;
    if (i < N_NODES) {
        float d = deg[i];
        deg[i] = d > 0.f ? rsqrtf(d) : 0.f;
    }
}

__global__ void k_norm(const int* __restrict__ src, const int* __restrict__ dst,
                       const float* __restrict__ w, const float* __restrict__ dis,
                       float* __restrict__ norm) {
    int e = blockIdx.x * 256 + threadIdx.x;
    if (e < N_EDGES) norm[e] = dis[src[e]] * w[e] * dis[dst[e]];
}

__global__ void k_cast16(const float* __restrict__ s, _Float16* __restrict__ d, int n) {
    int i = blockIdx.x * 256 + threadIdx.x;
    if (i < n) d[i] = (_Float16)s[i];
}

// --------------------------------------------------------- weight packing
// Pack a [64,64] f32 weight into WMMA B-operand layout:
//   dst[((kt*4+nt)*32 + lane)*16 + j] = W[32*kt+16*g+8*hi+jl][16*nt+lo]
__global__ void k_pack_w(const float* __restrict__ W, _Float16* __restrict__ dst) {
    int i = blockIdx.x * 256 + threadIdx.x;    // 4096 elements
    if (i >= HID * HID) return;
    int j    = i & 15;
    int lane = (i >> 4) & 31;
    int nt   = (i >> 9) & 3;
    int kt   = i >> 11;
    int lo = lane & 15, hi = lane >> 4;
    int g = j >> 3, jl = j & 7;
    int k = 32 * kt + 16 * g + 8 * hi + jl;
    int n = 16 * nt + lo;
    dst[i] = (_Float16)W[k * HID + n];
}

// Pack fc1_w [25600,256] f32 into per-(kstep,nt)-tile WMMA B layout:
//   dst[((ks*16+nt)*32 + lane)*16 + j] = B[32*ks+16*g+8*hi+jl][16*nt+lo]
__global__ void k_pack_fc1(const float* __restrict__ B, _Float16* __restrict__ dst) {
    int i = blockIdx.x * 256 + threadIdx.x;    // 6,553,600 elements
    if (i >= KFC1 * NFC1) return;
    int j    = i & 15;
    int lane = (i >> 4) & 31;
    int nt   = (i >> 9) & 15;
    int ks   = i >> 13;
    int lo = lane & 15, hi = lane >> 4;
    int g = j >> 3, jl = j & 7;
    int k = 32 * ks + 16 * g + 8 * hi + jl;
    int n = 16 * nt + lo;
    dst[i] = (_Float16)B[(size_t)k * NFC1 + n];
}

// ------------------------------------------------- node GEMM: [N,64]x[64,64]
// one wave -> 16 rows x 64 cols; packed W staged in LDS; 8 WMMAs per wave
__global__ __launch_bounds__(128) void k_gemm_node(const _Float16* __restrict__ A,
                                                   const _Float16* __restrict__ Wp,
                                                   float* __restrict__ out) {
    __shared__ _Float16 ldsW[HID * HID];       // packed layout, 8KB
    int tid = threadIdx.x;
    for (int i = tid; i < HID * HID / 8; i += 128)
        ((v8h*)ldsW)[i] = ((const v8h*)Wp)[i];
    __syncthreads();

    int wave = tid >> 5, lane = tid & 31;
    int lo = lane & 15, hi = lane >> 4;
    int rowTile = blockIdx.x * 4 + wave;       // 16-row tile index
    const _Float16* arow = A + (size_t)(rowTile * 16 + lo) * HID + 8 * hi;

    v16h a0 = hcombine(*(const v8h*)(arow),      *(const v8h*)(arow + 16));
    v16h a1 = hcombine(*(const v8h*)(arow + 32), *(const v8h*)(arow + 48));

    for (int nt = 0; nt < 4; ++nt) {
        const v8h* b0p = (const v8h*)(ldsW + ((0 * 4 + nt) * 32 + lane) * 16);
        const v8h* b1p = (const v8h*)(ldsW + ((1 * 4 + nt) * 32 + lane) * 16);
        v16h b0 = hcombine(b0p[0], b0p[1]);
        v16h b1 = hcombine(b1p[0], b1p[1]);
        v8f c = {};
        c = __builtin_amdgcn_wmma_f32_16x16x32_f16(false, a0, false, b0, (short)0, c, false, false);
        c = __builtin_amdgcn_wmma_f32_16x16x32_f16(false, a1, false, b1, (short)0, c, false, false);

        int rbase = rowTile * 16 + 8 * hi;
        for (int r = 0; r < 8; ++r)
            out[(size_t)(rbase + r) * HID + 16 * nt + lo] = c[r];
    }
}

// ------------------------------------------------------------- aggregation
__global__ void k_self(const float* __restrict__ hw, const float* __restrict__ dis,
                       float* __restrict__ out) {
    int i = blockIdx.x * 256 + threadIdx.x;
    if (i < N_NODES * HID) {
        float d = dis[i >> 6];
        out[i] = hw[i] * d * d;                // self-loop message (w=1)
    }
}

__global__ void k_scatter(const int* __restrict__ src, const int* __restrict__ dst,
                          const float* __restrict__ norm, const float* __restrict__ hw,
                          float* __restrict__ out) {
    int idx = blockIdx.x * 256 + threadIdx.x;
    if (idx >= N_EDGES * 16) return;
    int e = idx >> 4, part = idx & 15;
    int s = src[e], d = dst[e];
    float nm = norm[e];
    v4f v = *(const v4f*)(hw + (size_t)s * HID + part * 4);
    float* od = out + (size_t)d * HID + part * 4;
    atomicAdd(od + 0, v[0] * nm);
    atomicAdd(od + 1, v[1] * nm);
    atomicAdd(od + 2, v[2] * nm);
    atomicAdd(od + 3, v[3] * nm);
}

__global__ void k_bias_act_cast(const float* __restrict__ agg, const float* __restrict__ b,
                                _Float16* __restrict__ h16) {
    int i = blockIdx.x * 256 + threadIdx.x;
    if (i < N_NODES * HID) {
        float v = agg[i] + b[i & (HID - 1)];
        v = v >= 0.f ? v : NEG_SLOPE * v;
        h16[i] = (_Float16)v;
    }
}

// --------------------------------------------- fc1 GEMM: [128,25600]x[25600,256]
// split-K x8: 256 blocks x 4 waves = 1024 waves; each wave: one 16x16 tile,
// 100 K-steps, f32 atomic reduction into bias-initialized output.
__global__ void k_bias_init(const float* __restrict__ bias, float* __restrict__ g1) {
    int i = blockIdx.x * 256 + threadIdx.x;
    if (i < N_GRAPHS * NFC1) g1[i] = bias[i & (NFC1 - 1)];
}

__global__ __launch_bounds__(128) void k_gemm_fc1(const _Float16* __restrict__ A,
                                                  const _Float16* __restrict__ Bp,
                                                  float* __restrict__ out) {
    int tid = threadIdx.x, wave = tid >> 5, lane = tid & 31;
    int lo = lane & 15, hi = lane >> 4;
    int tilegrp = blockIdx.x & 31;             // 32 groups of 4 tiles
    int ksplit  = blockIdx.x >> 5;             // 0..7
    int tile = tilegrp * 4 + wave;             // 0..127
    int mt = tile >> 4, nt = tile & 15;

    const _Float16* arow = A + (size_t)(mt * 16 + lo) * KFC1 + 8 * hi;
    int k0 = ksplit * (KSTEPS / KSPLIT);

    v8f c = {};
    for (int k = k0; k < k0 + KSTEPS / KSPLIT; ++k) {
        const v8h* ap = (const v8h*)(arow + k * 32);
        v16h a = hcombine(ap[0], *(const v8h*)(arow + k * 32 + 16));
        const v8h* bp = (const v8h*)(Bp + (size_t)((k * 16 + nt) * 32 + lane) * 16);
        v16h b = hcombine(bp[0], bp[1]);
        c = __builtin_amdgcn_wmma_f32_16x16x32_f16(false, a, false, b, (short)0, c, false, false);
    }

    int col = nt * 16 + lo;
    int rbase = mt * 16 + 8 * hi;
    for (int r = 0; r < 8; ++r)
        atomicAdd(&out[(size_t)(rbase + r) * NFC1 + col], c[r]);
}

// ------------------------------------------------------------------ small FCs
__global__ void k_fc2(const float* __restrict__ g1, const float* __restrict__ w,
                      const float* __restrict__ b, float* __restrict__ out) {
    int i = blockIdx.x * 256 + threadIdx.x;
    if (i >= N_GRAPHS * 128) return;
    int g = i >> 7, c = i & 127;
    float s = b[c];
    for (int k = 0; k < 256; ++k) s += g1[g * 256 + k] * w[k * 128 + c];
    out[i] = s;
}

__global__ void k_fc3(const float* __restrict__ g2, const float* __restrict__ w,
                      const float* __restrict__ b, float* __restrict__ out) {
    int i = blockIdx.x * 256 + threadIdx.x;
    if (i >= N_GRAPHS * 2) return;
    int g = i >> 1, c = i & 1;
    float s = b[c];
    for (int k = 0; k < 128; ++k) s += g2[g * 128 + k] * w[k * 2 + c];
    out[i] = s;
}

__global__ void k_penal(const float* __restrict__ logits, const float* __restrict__ logits0,
                        float* __restrict__ out) {
    __shared__ float red[256];
    int t = threadIdx.x;
    float d = logits[t] - logits0[t];
    red[t] = d * d;
    __syncthreads();
    for (int s = 128; s > 0; s >>= 1) {
        if (t < s) red[t] += red[t + s];
        __syncthreads();
    }
    if (t == 0) out[N_GRAPHS * 2] = 0.09f * (float)N_GRAPHS / sqrtf(red[0]);
}

// ---------------------------------------------------------------------------
extern "C" void kernel_launch(void* const* d_in, const int* in_sizes, int n_in,
                              void* d_out, int out_size, void* d_ws, size_t ws_size,
                              hipStream_t stream) {
    const float* x    = (const float*)d_in[0];
    const int*   ei   = (const int*)d_in[1];
    const int*   esrc = ei;
    const int*   edst = ei + N_EDGES;
    const float* ew   = (const float*)d_in[2];
    const float* Wl[4] = {(const float*)d_in[3], (const float*)d_in[5],
                          (const float*)d_in[7], (const float*)d_in[9]};
    const float* bl[4] = {(const float*)d_in[4], (const float*)d_in[6],
                          (const float*)d_in[8], (const float*)d_in[10]};
    const float* fc1_w = (const float*)d_in[11];
    const float* fc1_b = (const float*)d_in[12];
    const float* fc2_w = (const float*)d_in[13];
    const float* fc2_b = (const float*)d_in[14];
    const float* fc3_w = (const float*)d_in[15];
    const float* fc3_b = (const float*)d_in[16];
    float* outp = (float*)d_out;

    // workspace carve-up (256B aligned slices)
    size_t off = 0;
    auto alloc = [&](size_t bytes) -> void* {
        void* p = (char*)d_ws + off;
        off += (bytes + 255) & ~(size_t)255;
        return p;
    };
    float*    deg    = (float*)alloc((size_t)N_NODES * 4);            // becomes dis
    float*    norm   = (float*)alloc((size_t)N_EDGES * 4);
    float*    hw     = (float*)alloc((size_t)N_NODES * HID * 4);
    float*    hcur   = (float*)alloc((size_t)N_NODES * HID * 4);
    _Float16* x16    = (_Float16*)alloc((size_t)N_NODES * HID * 2);
    _Float16* h16    = (_Float16*)alloc((size_t)N_NODES * HID * 2);
    _Float16* wp16   = (_Float16*)alloc((size_t)4 * HID * HID * 2);   // packed
    _Float16* fc1wp  = (_Float16*)alloc((size_t)KFC1 * NFC1 * 2);     // packed
    float*    g1     = (float*)alloc((size_t)N_GRAPHS * 256 * 4);
    float*    g1x    = (float*)alloc((size_t)N_GRAPHS * 256 * 4);
    float*    g2     = (float*)alloc((size_t)N_GRAPHS * 128 * 4);
    float*    g2x    = (float*)alloc((size_t)N_GRAPHS * 128 * 4);
    float*    lg0    = (float*)alloc((size_t)N_GRAPHS * 2 * 4);

    const int nodeB = (N_NODES + 255) / 256;               // 200
    const int edgeB = (N_EDGES + 255) / 256;               // 6400
    const int nhB   = (N_NODES * HID + 255) / 256;         // 12800

    // GCN normalization
    k_deg_init<<<nodeB, 256, 0, stream>>>(deg);
    k_deg_edges<<<edgeB, 256, 0, stream>>>(edst, ew, deg);
    k_dis<<<nodeB, 256, 0, stream>>>(deg);
    k_norm<<<edgeB, 256, 0, stream>>>(esrc, edst, ew, deg, norm);

    // f16 casts + WMMA-layout weight packing
    k_cast16<<<nhB, 256, 0, stream>>>(x, x16, N_NODES * HID);
    for (int l = 0; l < 4; ++l)
        k_pack_w<<<16, 256, 0, stream>>>(Wl[l], wp16 + l * HID * HID);
    k_pack_fc1<<<(KFC1 * NFC1 + 255) / 256, 256, 0, stream>>>(fc1_w, fc1wp);

    // 4 GCN layers
    for (int l = 0; l < 4; ++l) {
        const _Float16* Ain = (l == 0) ? x16 : h16;
        k_gemm_node<<<N_NODES / 64, 128, 0, stream>>>(Ain, wp16 + l * HID * HID, hw);
        k_self<<<nhB, 256, 0, stream>>>(hw, deg, hcur);
        k_scatter<<<(N_EDGES * 16) / 256, 256, 0, stream>>>(esrc, edst, norm, hw, hcur);
        k_bias_act_cast<<<nhB, 256, 0, stream>>>(hcur, bl[l], h16);
    }

    // head on h
    k_bias_init<<<(N_GRAPHS * NFC1 + 255) / 256, 256, 0, stream>>>(fc1_b, g1);
    k_gemm_fc1<<<32 * KSPLIT, 128, 0, stream>>>(h16, fc1wp, g1);
    k_fc2<<<(N_GRAPHS * 128) / 256, 256, 0, stream>>>(g1, fc2_w, fc2_b, g2);
    k_fc3<<<1, 256, 0, stream>>>(g2, fc3_w, fc3_b, outp);

    // head on x (for logits0)
    k_bias_init<<<(N_GRAPHS * NFC1 + 255) / 256, 256, 0, stream>>>(fc1_b, g1x);
    k_gemm_fc1<<<32 * KSPLIT, 128, 0, stream>>>(x16, fc1wp, g1x);
    k_fc2<<<(N_GRAPHS * 128) / 256, 256, 0, stream>>>(g1x, fc2_w, fc2_b, g2x);
    k_fc3<<<1, 256, 0, stream>>>(g2x, fc3_w, fc3_b, lg0);

    // penalty scalar -> out[256]
    k_penal<<<1, 256, 0, stream>>>(outp, lg0, outp);
}
